// MaskedMultiHeadAttention_33629593927878
// MI455X (gfx1250) — compile-verified, matrix operand order checked by reading, not run on hardware
//
#include <hip/hip_runtime.h>

#define D_MODEL   512
#define NUM_HEADS 8
#define HEAD_DIM  64
#define BATCH     2
#define SEQ       2048

typedef __attribute__((ext_vector_type(16))) _Float16 v16h;
typedef __attribute__((ext_vector_type(8)))  _Float16 v8h;
typedef __attribute__((ext_vector_type(8)))  float    v8f;
typedef __attribute__((ext_vector_type(4)))  float    v4f;

union AB16 { v16h v; v8h h[2]; };

// convert 8 consecutive f32 -> v8h (two 16B loads + cvt)
__device__ __forceinline__ v8h cvt8(const float* __restrict__ p) {
    const v4f f0 = *(const v4f*)p;
    const v4f f1 = *(const v4f*)(p + 4);
    v8h r;
#pragma unroll
    for (int i = 0; i < 4; ++i) {
        r[i]     = (_Float16)f0[i];
        r[i + 4] = (_Float16)f1[i];
    }
    return r;
}

__device__ __forceinline__ v8f wmma16(v16h a, v16h b, v8f c) {
    return __builtin_amdgcn_wmma_f32_16x16x32_f16(false, a, false, b,
                                                  (short)0, c, false, false);
}

// ---------------------------------------------------------------------------
// Fused projection: P = relu(X @ W^T + b), emitted as f16.
//   z==0 -> Qh [b,h,s,dh]   z==1 -> Kh [b,h,s,dh]   z==2 -> Vt [b,h,dh,s]
// Grid: (M/128, 512/64, 3), block 256 (8 waves). Wave computes 16(M) x 64(N).
// ---------------------------------------------------------------------------
__global__ __launch_bounds__(256)
void proj_kernel(const float* __restrict__ Xq, const float* __restrict__ Xk,
                 const float* __restrict__ Xv, const float* __restrict__ W,
                 const float* __restrict__ bias,
                 _Float16* __restrict__ Qh, _Float16* __restrict__ Kh,
                 _Float16* __restrict__ Vt) {
    __shared__ __align__(16) _Float16 lds_w[64 * 32];    // W tile (B^T rows)
    __shared__ __align__(16) _Float16 lds_c[8][16 * 64]; // per-wave store staging

    const int tid  = threadIdx.x;
    const int wave = tid >> 5;
    const int lane = tid & 31;
    const int grp  = lane >> 4;   // half-wave group (K-group of WMMA layouts)
    const int l16  = lane & 15;

    const int z = blockIdx.z;
    const float* X = (z == 0) ? Xq : (z == 1) ? Xk : Xv;

    const int m0 = blockIdx.x * 128 + wave * 16;  // first M row of this wave
    const int n0 = blockIdx.y * 64;               // first N col (== head*64)

    v8f acc[4] = {};

    for (int kk = 0; kk < D_MODEL; kk += 32) {
        // cooperative W tile: rows e = n0..n0+63, cols kk..kk+31 (f32 -> f16)
        {
            const int e  = tid >> 2;        // 0..63
            const int k8 = (tid & 3) * 8;   // 0,8,16,24
            *(v8h*)(lds_w + e * 32 + k8) =
                cvt8(W + (size_t)(n0 + e) * D_MODEL + kk + k8);
        }
        // A operand: 16 rows x 32 K of X, per-lane WMMA A layout
        AB16 a;
        {
            const float* xr = X + (size_t)(m0 + l16) * D_MODEL + kk;
            a.h[0] = cvt8(xr + grp * 8);
            a.h[1] = cvt8(xr + 16 + grp * 8);
            if (kk + 32 < D_MODEL)
                __builtin_prefetch(xr + 32, 0, 3);   // global_prefetch next k slice
        }
        __syncthreads();
#pragma unroll
        for (int j = 0; j < 4; ++j) {
            AB16 bm;
            const _Float16* bp = lds_w + (j * 16 + l16) * 32;
            bm.h[0] = *(const v8h*)(bp + grp * 8);
            bm.h[1] = *(const v8h*)(bp + 16 + grp * 8);
            acc[j] = wmma16(a.v, bm.v, acc[j]);
        }
        __syncthreads();
    }

    // ---- epilogue: + bias, relu, f16, packed 128-bit stores ----
    const int h  = blockIdx.y;        // one head per N tile
    const int b  = m0 >> 11;          // whole 16-row tile is in one batch
    const int s0 = m0 & (SEQ - 1);

    if (z == 2) {
        // Vt [dh][s]: C-layout register column r = consecutive s -> direct v8h
#pragma unroll
        for (int j = 0; j < 4; ++j) {
            const int dh = j * 16 + l16;
            const float bv = bias[n0 + dh];
            v8h pk;
#pragma unroll
            for (int r = 0; r < 8; ++r) {
                float val = acc[j][r] + bv;
                val = val > 0.f ? val : 0.f;
                pk[r] = (_Float16)val;
            }
            *(v8h*)(Vt + ((size_t)(b * NUM_HEADS + h) * HEAD_DIM + dh) * SEQ +
                    s0 + grp * 8) = pk;
        }
    } else {
        // Qh/Kh [s][dh]: stage tile in LDS, then coalesced b128 copy-out
        _Float16* dst = (z == 0) ? Qh : Kh;
        _Float16* cw  = lds_c[wave];
#pragma unroll
        for (int j = 0; j < 4; ++j) {
            const float bv = bias[n0 + j * 16 + l16];
#pragma unroll
            for (int r = 0; r < 8; ++r) {
                float val = acc[j][r] + bv;
                val = val > 0.f ? val : 0.f;
                cw[(grp * 8 + r) * 64 + j * 16 + l16] = (_Float16)val;
            }
        }
        // same-wave DS ordering; 16 rows x 8 segs = 128 chunks, 4 per lane
#pragma unroll
        for (int i = 0; i < 4; ++i) {
            const int c   = lane + 32 * i;
            const int row = c >> 3, seg = c & 7;
            const v8h pk = *(const v8h*)(cw + row * 64 + seg * 8);
            *(v8h*)(dst + ((size_t)(b * NUM_HEADS + h) * SEQ + s0 + row) *
                    HEAD_DIM + seg * 8) = pk;
        }
    }
}

// ---------------------------------------------------------------------------
// Causal flash attention. Grid: (SEQ/128, B*H), block 256 (8 waves).
// Wave owns 16 q rows; KV tiles of 64 staged in LDS, shared by all waves.
// 16 WMMAs per wave per KV tile between one barrier pair.
// ---------------------------------------------------------------------------
__global__ __launch_bounds__(256)
void attn_kernel(const _Float16* __restrict__ Qh, const _Float16* __restrict__ Kh,
                 const _Float16* __restrict__ Vt, float* __restrict__ out) {
    // 32 KB overlay: [K 8KB | V 8KB | P 16KB] during loop, O staging after
    __shared__ __align__(16) unsigned char smem[32 * 1024];
    _Float16* lds_k = (_Float16*)smem;                 // [kv 64][d 64]
    _Float16* lds_v = (_Float16*)(smem + 8 * 1024);    // [d 64][kv 64]
    _Float16* lds_p = (_Float16*)(smem + 16 * 1024);   // 8 waves x [16][64]
    float*    lds_o = (float*)smem;                    // 8 waves x [16][64] f32

    const int tid  = threadIdx.x;
    const int wave = tid >> 5;
    const int lane = tid & 31;
    const int grp  = lane >> 4;
    const int l16  = lane & 15;

    const int bh = blockIdx.y;                 // b*H + h
    const int q0 = blockIdx.x * 128;
    const int qr = q0 + wave * 16;             // wave's first query row
    const int qr_u = __builtin_amdgcn_readfirstlane(qr);  // uniform branch key

    const _Float16* Qb = Qh + (size_t)bh * SEQ * HEAD_DIM;
    const _Float16* Kb = Kh + (size_t)bh * SEQ * HEAD_DIM;
    const _Float16* Vb = Vt + (size_t)bh * HEAD_DIM * SEQ;

    // Q A-operands for K-steps d=0..31 and d=32..63
    AB16 qa0, qa1;
    {
        const _Float16* qp = Qb + (size_t)(qr + l16) * HEAD_DIM;
        qa0.h[0] = *(const v8h*)(qp + grp * 8);
        qa0.h[1] = *(const v8h*)(qp + 16 + grp * 8);
        qa1.h[0] = *(const v8h*)(qp + 32 + grp * 8);
        qa1.h[1] = *(const v8h*)(qp + 48 + grp * 8);
    }

    float m_run[8], l_run[8];
    v8f o[4] = {};
#pragma unroll
    for (int r = 0; r < 8; ++r) { m_run[r] = -__builtin_inff(); l_run[r] = 0.f; }

    const float scale = 0.125f;  // 1/sqrt(64)
    const int kv_end = q0 + 128;

    for (int kv0 = 0; kv0 < kv_end; kv0 += 64) {
        // ---- cooperative LDS fill: K 64x64 + V^T 64x64 halves, 2 chunks ea ----
#pragma unroll
        for (int i = 0; i < 2; ++i) {
            const int c  = tid + 256 * i;     // 0..511
            const int rw = c >> 3, sg = c & 7;
            *(v8h*)(lds_k + rw * 64 + sg * 8) =
                *(const v8h*)(Kb + (size_t)(kv0 + rw) * HEAD_DIM + sg * 8);
            *(v8h*)(lds_v + rw * 64 + sg * 8) =
                *(const v8h*)(Vb + (size_t)rw * SEQ + kv0 + sg * 8);
        }
        if (kv0 + 64 < kv_end) {   // prefetch next KV tile into L2/L0
            __builtin_prefetch(Kb + (size_t)(kv0 + 64 + (tid & 63)) * HEAD_DIM, 0, 3);
            __builtin_prefetch(Vb + (size_t)(tid & 63) * SEQ + kv0 + 64, 0, 3);
        }
        __syncthreads();

        if (kv0 <= qr_u + 15) {  // wave-uniform causal skip
            // ---- S = Q K^T  (4 col-tiles x 2 k-steps) ----
            v8f sc[4];
#pragma unroll
            for (int j = 0; j < 4; ++j) {
                AB16 bk0, bk1;
                const _Float16* kp = lds_k + (j * 16 + l16) * 64;
                bk0.h[0] = *(const v8h*)(kp + grp * 8);
                bk0.h[1] = *(const v8h*)(kp + 16 + grp * 8);
                bk1.h[0] = *(const v8h*)(kp + 32 + grp * 8);
                bk1.h[1] = *(const v8h*)(kp + 48 + grp * 8);
                v8f c = {};
                c = wmma16(qa0.v, bk0.v, c);
                c = wmma16(qa1.v, bk1.v, c);
                sc[j] = c;
            }
            // ---- mask + scale + row max ----
            float mt[8];
#pragma unroll
            for (int r = 0; r < 8; ++r) {
                const int srow = qr + grp * 8 + r;
                float mx = -__builtin_inff();
#pragma unroll
                for (int j = 0; j < 4; ++j) {
                    float s = sc[j][r] * scale;
                    if (kv0 + j * 16 + l16 > srow) s = -__builtin_inff();
                    sc[j][r] = s;
                    mx = fmaxf(mx, s);
                }
                mt[r] = mx;
            }
#pragma unroll
            for (int r = 0; r < 8; ++r)
#pragma unroll
                for (int mk = 1; mk < 16; mk <<= 1)
                    mt[r] = fmaxf(mt[r], __shfl_xor(mt[r], mk, 32));
            // ---- online softmax update ----
            float alpha[8], rs[8];
#pragma unroll
            for (int r = 0; r < 8; ++r) {
                const float mn = fmaxf(m_run[r], mt[r]);
                alpha[r] = __expf(m_run[r] - mn);
                m_run[r] = mn;
                float acc = 0.f;
#pragma unroll
                for (int j = 0; j < 4; ++j) {
                    const float p = __expf(sc[j][r] - mn);
                    sc[j][r] = p;
                    acc += p;
                }
                rs[r] = acc;
            }
#pragma unroll
            for (int r = 0; r < 8; ++r)
#pragma unroll
                for (int mk = 1; mk < 16; mk <<= 1)
                    rs[r] += __shfl_xor(rs[r], mk, 32);
#pragma unroll
            for (int r = 0; r < 8; ++r) l_run[r] = l_run[r] * alpha[r] + rs[r];
#pragma unroll
            for (int t = 0; t < 4; ++t)
#pragma unroll
                for (int r = 0; r < 8; ++r) o[t][r] *= alpha[r];

            // ---- P: C layout -> LDS -> A layout (same-wave DS ordering) ----
            _Float16* pw = lds_p + wave * (16 * 64);
#pragma unroll
            for (int j = 0; j < 4; ++j)
#pragma unroll
                for (int r = 0; r < 8; ++r)
                    pw[(grp * 8 + r) * 64 + j * 16 + l16] = (_Float16)sc[j][r];
            AB16 pa0, pa1;
            {
                const _Float16* pp = pw + l16 * 64;
                pa0.h[0] = *(const v8h*)(pp + grp * 8);
                pa0.h[1] = *(const v8h*)(pp + 16 + grp * 8);
                pa1.h[0] = *(const v8h*)(pp + 32 + grp * 8);
                pa1.h[1] = *(const v8h*)(pp + 48 + grp * 8);
            }
            // ---- O += P V  (4 dh-tiles x 2 k-steps) ----
#pragma unroll
            for (int t = 0; t < 4; ++t) {
                AB16 bv0, bv1;
                const _Float16* vp = lds_v + (t * 16 + l16) * 64;
                bv0.h[0] = *(const v8h*)(vp + grp * 8);
                bv0.h[1] = *(const v8h*)(vp + 16 + grp * 8);
                bv1.h[0] = *(const v8h*)(vp + 32 + grp * 8);
                bv1.h[1] = *(const v8h*)(vp + 48 + grp * 8);
                o[t] = wmma16(pa0.v, bv0.v, o[t]);
                o[t] = wmma16(pa1.v, bv1.v, o[t]);
            }
        }
        __syncthreads();
    }

    // ---- normalize, stage f32 tile in LDS overlay, coalesced b128 stores ----
    const int b = bh / NUM_HEADS;
    const int h = bh % NUM_HEADS;
    float inv_l[8];
#pragma unroll
    for (int r = 0; r < 8; ++r) inv_l[r] = 1.0f / l_run[r];

    float* ow = lds_o + wave * (16 * 64);   // safe: loop ended with barrier
#pragma unroll
    for (int t = 0; t < 4; ++t)
#pragma unroll
        for (int r = 0; r < 8; ++r)
            ow[(grp * 8 + r) * 64 + t * 16 + l16] = o[t][r] * inv_l[r];
    // 16 rows x 16 segs (v4f) = 256 chunks, 8 per lane; same-wave DS ordering
#pragma unroll
    for (int i = 0; i < 8; ++i) {
        const int c   = lane + 32 * i;
        const int row = c >> 4, seg = c & 15;
        const v4f val = *(const v4f*)(ow + row * 64 + seg * 4);
        *(v4f*)(out + ((size_t)(h * BATCH + b) * SEQ + qr + row) * HEAD_DIM +
                seg * 4) = val;
    }
}

// ---------------------------------------------------------------------------
extern "C" void kernel_launch(void* const* d_in, const int* in_sizes, int n_in,
                              void* d_out, int out_size, void* d_ws, size_t ws_size,
                              hipStream_t stream) {
    const float* q    = (const float*)d_in[0];
    const float* k    = (const float*)d_in[1];
    const float* v    = (const float*)d_in[2];
    const float* W    = (const float*)d_in[3];
    const float* bias = (const float*)d_in[4];

    const size_t per = (size_t)BATCH * NUM_HEADS * SEQ * HEAD_DIM;  // 2,097,152
    _Float16* Qh = (_Float16*)d_ws;
    _Float16* Kh = Qh + per;
    _Float16* Vt = Kh + per;

    dim3 pgrid((BATCH * SEQ) / 128, D_MODEL / 64, 3);
    proj_kernel<<<pgrid, 256, 0, stream>>>(q, k, v, W, bias, Qh, Kh, Vt);

    dim3 agrid(SEQ / 128, BATCH * NUM_HEADS);
    attn_kernel<<<agrid, 256, 0, stream>>>(Qh, Kh, Vt, (float*)d_out);
}
